// LocalScaledDotProdAttV1_9028021256463
// MI455X (gfx1250) — matrix operand order repaired and checked by reading, not compiled, not run on hardware
//
#include <hip/hip_runtime.h>
#include <hip/hip_bf16.h>

// ---------------- problem constants ----------------
#define B_   8
#define T_   4000
#define F_   512          // IN_FEATS = OUT_FEATS = H*DK
#define H_   8
#define DK_  64
#define TP   4032         // padded T: nb=63 blocks of 64
#define NB1  63
#define NB2  62
#define NEGV (-1e30f)

typedef __attribute__((ext_vector_type(16))) __bf16 v16bf;
typedef __attribute__((ext_vector_type(8)))  float  v8f;

// float <-> bf16 helpers (RNE)
__device__ __forceinline__ unsigned short bfbits(float f) {
  unsigned u = __float_as_uint(f);
  return (unsigned short)((u + 0x7FFFu + ((u >> 16) & 1u)) >> 16);
}
__device__ __forceinline__ __bf16 f2bf(float f) {
  unsigned short s = bfbits(f);
  return *(__bf16*)&s;
}
__device__ __forceinline__ unsigned pk2(float a, float b) {
  return (unsigned)bfbits(a) | ((unsigned)bfbits(b) << 16);
}
__device__ __forceinline__ float bf2f_lo(unsigned w) { return __uint_as_float(w << 16); }
__device__ __forceinline__ float bf2f_hi(unsigned w) { return __uint_as_float(w & 0xFFFF0000u); }

// ---- WMMA bf16 16x16x32 fragment loaders (contiguous k order, same in A and B) ----
// A (16x32, MxK): lane = half*16 + r; row M=r; K chunk [16*half .. 16*half+15]
__device__ __forceinline__ v16bf frag_a_ld(const __bf16* rowbase, int pitch, int lane) {
  int half = (lane >> 4) & 1;
  int r    = lane & 15;
  const __bf16* p = rowbase + r * pitch + 16 * half;
  v16bf f;
  ((uint4*)&f)[0] = *(const uint4*)(p);
  ((uint4*)&f)[1] = *(const uint4*)(p + 8);
  return f;
}
// B (32x16, KxN) stored N-major (each "row" = one output column, K contiguous):
// lane = half*16 + c; col N=c; K chunk [16*half .. 16*half+15]
__device__ __forceinline__ v16bf frag_b_ld(const __bf16* colbase, int pitch, int lane) {
  int half = (lane >> 4) & 1;
  int c    = lane & 15;
  const __bf16* p = colbase + c * pitch + 16 * half;
  v16bf f;
  ((uint4*)&f)[0] = *(const uint4*)(p);
  ((uint4*)&f)[1] = *(const uint4*)(p + 8);
  return f;
}

__device__ __forceinline__ v8f wmma_bf16(v16bf a, v16bf b, v8f c) {
  return __builtin_amdgcn_wmma_f32_16x16x32_bf16(false, a, false, b, (short)0, c, false, false);
}

// ---------------- kernel 0: weights fp32 -> bf16, packed in B-fragment order ----
// out[((nTile*16 + ks)*32 + lane)*16 + e] = bf16( W[k*512 + n] )
//   n = nTile*16 + (lane&15);  k = ks*32 + 16*(lane>>4) + e
__global__ __launch_bounds__(256) void pack_weights(
    const float* __restrict__ w0, const float* __restrict__ w1,
    const float* __restrict__ w2, const float* __restrict__ w3,
    __bf16* __restrict__ out) {
  int i = blockIdx.x * 256 + threadIdx.x;      // 4 * 262144 total
  int sel   = i >> 18;
  int r     = i & 0x3FFFF;
  int e     = r & 15;
  int lane  = (r >> 4) & 31;
  int ks    = (r >> 9) & 15;
  int nTile = (r >> 13) & 31;
  int n = nTile * 16 + (lane & 15);
  int k = ks * 32 + 16 * (lane >> 4) + e;
  const float* s = (sel == 0) ? w0 : (sel == 1) ? w1 : (sel == 2) ? w2 : w3;
  out[i] = f2bf(s[k * F_ + n]);
}

// ---------------- kernel: zero rows [0,32) of each batch in x2 (pass-2 gap) ----
__global__ __launch_bounds__(256) void zero_x2_head(__bf16* __restrict__ x2) {
  int i = blockIdx.x * 256 + threadIdx.x;      // 65536 uint writes (8*32 rows * 256)
  int col2   = i & 255;                        // 256 uints per 512-col row
  int rowIdx = i >> 8;                         // 0..255
  int b = rowIdx >> 5, t = rowIdx & 31;
  ((unsigned*)x2)[((size_t)b * TP + t) * 256 + col2] = 0u;
}

// ---------------- GEMM: 64(M) x 512(N) per workgroup, K=512, bf16 WMMA -------
// A read from HBM exactly once (grid.y == 1); B from L2-resident fragment-packed
// weights via coalesced global_load_b128 (no LDS for B).
// MODE 0: A = fp32 activations (B x T_ x F_) -> padded rows zero; Out bf16 (B x TP x F_).
// MODE 1: A = x1 + x2 (two bf16 buffers, B x TP x F_) at rows t < T_; Out fp32 (B x T_ x F_).
template <int MODE>
__global__ __launch_bounds__(256) void gemm_wmma(
    const void*   __restrict__ A0,
    const void*   __restrict__ A1,      // MODE1 only: second bf16 buffer
    const __bf16* __restrict__ Wp,      // fragment-packed 512x512
    const float*  __restrict__ bias,    // 512
    void*         __restrict__ Out) {
  __shared__ __align__(16) __bf16 sA[64 * 40];

  const int tid  = threadIdx.x;
  const int lane = tid & 31;
  const int wave = tid >> 5;            // 8 waves: 2(M) x 4(N)
  const int wM   = wave & 1;            // 2 x 32 rows
  const int wN   = wave >> 1;           // 4 x 128 cols
  const int mBlk = blockIdx.x;

  v8f acc[2][8];
#pragma unroll
  for (int i = 0; i < 2; ++i)
#pragma unroll
    for (int j = 0; j < 8; ++j)
      acc[i][j] = (v8f){0.f, 0.f, 0.f, 0.f, 0.f, 0.f, 0.f, 0.f};

  // per-thread A staging coords: one row, 8 consecutive k
  const int arow = tid >> 2;            // 0..63
  const int ak8  = (tid & 3) * 8;       // 0,8,16,24

  for (int ks = 0; ks < 16; ++ks) {
    const int kb = ks * 32;
    // ---- stage A: 64x32 -> bf16 (packed b128 store), M-major, pitch 40 ----
    {
      int m = mBlk * 64 + arow;
      uint4 h;
      if (MODE == 0) {
        int batch = m / TP, t = m % TP;
        if (t < T_) {
          const float* p = (const float*)A0 + ((size_t)batch * T_ + t) * F_ + kb + ak8;
          float4 v0 = *(const float4*)(p);
          float4 v1 = *(const float4*)(p + 4);
          h.x = pk2(v0.x, v0.y); h.y = pk2(v0.z, v0.w);
          h.z = pk2(v1.x, v1.y); h.w = pk2(v1.z, v1.w);
        } else {
          h = make_uint4(0u, 0u, 0u, 0u);     // jnp.pad rows
        }
      } else {
        int batch = m / T_, t = m % T_;
        size_t idx = ((size_t)batch * TP + t) * F_ + kb + ak8;
        uint4 a = *(const uint4*)((const __bf16*)A0 + idx);
        uint4 b = *(const uint4*)((const __bf16*)A1 + idx);
        const unsigned* ap = (const unsigned*)&a;
        const unsigned* bp = (const unsigned*)&b;
        unsigned* hp = (unsigned*)&h;
#pragma unroll
        for (int q = 0; q < 4; ++q) {
          float lo = bf2f_lo(ap[q]) + bf2f_lo(bp[q]);
          float hi = __uint_as_float(ap[q] & 0xFFFF0000u) + __uint_as_float(bp[q] & 0xFFFF0000u);
          hp[q] = pk2(lo, hi);
        }
      }
      *(uint4*)(sA + arow * 40 + ak8) = h;
    }
    __syncthreads();

    v16bf af0 = frag_a_ld(sA + (wM * 32 + 0)  * 40, 40, lane);
    v16bf af1 = frag_a_ld(sA + (wM * 32 + 16) * 40, 40, lane);
    // B fragments straight from packed global (L2-hot), fully lane-coalesced
    const __bf16* wb = Wp + ((((wN * 8) * 16 + ks) << 9) + (lane << 4));
#pragma unroll
    for (int j = 0; j < 8; ++j) {
      v16bf bfr;
      ((uint4*)&bfr)[0] = *(const uint4*)(wb);
      ((uint4*)&bfr)[1] = *(const uint4*)(wb + 8);
      wb += (16 << 9);                  // next nTile
      acc[0][j] = wmma_bf16(af0, bfr, acc[0][j]);
      acc[1][j] = wmma_bf16(af1, bfr, acc[1][j]);
    }
    __syncthreads();
  }

  // ---- epilogue: bias + store ----
  const int rlo = ((lane >> 4) & 1) * 8;   // C/D: row = v + 8*half
  const int cl  = lane & 15;
#pragma unroll
  for (int i = 0; i < 2; ++i)
#pragma unroll
    for (int j = 0; j < 8; ++j) {
      int colg = wN * 128 + j * 16 + cl;
      float bb = bias[colg];
#pragma unroll
      for (int v = 0; v < 8; ++v) {
        int m = mBlk * 64 + wM * 32 + i * 16 + rlo + v;
        float val = acc[i][j][v] + bb;
        if (MODE == 0) {
          int t = m % TP;
          if (t >= T_) val = 0.f;       // q0p/k0p/v0p pad rows exactly zero
          ((__bf16*)Out)[(size_t)m * F_ + colg] = f2bf(val);
        } else {
          ((float*)Out)[(size_t)m * F_ + colg] = val;
        }
      }
    }
}

// ---------------- attention: one (b, h, 64-block, pass) per workgroup --------
// Writes bf16 output to xout (pass-specific buffer; no RMW).
__global__ __launch_bounds__(256) void attn_block(
    const __bf16* __restrict__ qbf, const __bf16* __restrict__ kbf,
    const __bf16* __restrict__ vbf, const unsigned char* __restrict__ mask,
    __bf16* __restrict__ xout, int pass) {
  __shared__ __align__(16) float  sS[64 * 65];
  __shared__ __align__(16) __bf16 sP[64 * 72];
  __shared__ __align__(16) __bf16 sVT[64 * 72];

  const int tid  = threadIdx.x;
  const int lane = tid & 31;
  const int wave = tid >> 5;
  const int half = (lane >> 4) & 1;
  const int cl   = lane & 15;
  const int blk  = blockIdx.x;
  const int h    = blockIdx.y;
  const int b    = blockIdx.z;
  const int r0   = pass ? (32 + blk * 64) : (blk * 64);   // padded-seq row start
  const size_t base = ((size_t)b * TP + r0) * F_ + (size_t)h * DK_;

  // wave's tiles: (ti, tj0) and (ti, tj0+1)
  const int ti  = wave >> 1;
  const int tj0 = (wave & 1) * 2;

  // stage V^T: sVT[d][k] = V[k][d]
  {
    int k  = tid >> 2;              // 0..63
    int d0 = (tid & 3) * 16;
    const __bf16* vp = vbf + base + (size_t)k * F_ + d0;
#pragma unroll
    for (int i = 0; i < 16; ++i) sVT[(d0 + i) * 72 + k] = vp[i];
  }

  // S = (Q K^T) / 8 ; Q fragment shared by both tj tiles
  v8f sc[2];
  sc[0] = (v8f){0.f,0.f,0.f,0.f,0.f,0.f,0.f,0.f};
  sc[1] = (v8f){0.f,0.f,0.f,0.f,0.f,0.f,0.f,0.f};
#pragma unroll
  for (int ds = 0; ds < 2; ++ds) {
    v16bf qa  = frag_a_ld(qbf + base + (size_t)(ti * 16) * F_ + ds * 32, F_, lane);
    v16bf kb0 = frag_b_ld(kbf + base + (size_t)(tj0 * 16) * F_ + ds * 32, F_, lane);
    v16bf kb1 = frag_b_ld(kbf + base + (size_t)((tj0 + 1) * 16) * F_ + ds * 32, F_, lane);
    sc[0] = wmma_bf16(qa, kb0, sc[0]);
    sc[1] = wmma_bf16(qa, kb1, sc[1]);
  }
#pragma unroll
  for (int u = 0; u < 2; ++u)
#pragma unroll
    for (int v = 0; v < 8; ++v)
      sS[(ti * 16 + half * 8 + v) * 65 + (tj0 + u) * 16 + cl] = sc[u][v] * 0.125f;
  __syncthreads();

  // masked softmax per row (threads 0..63), matches where->softmax->where
  if (tid < 64) {
    const int i  = tid;
    const int qi = r0 + i;
    const bool rowvalid = qi < T_;
    float mx = NEGV;
    for (int j = 0; j < 64; ++j) {
      int kj = r0 + j;
      bool valid = rowvalid && (kj < T_) &&
                   (mask[(size_t)b * T_ * T_ + (size_t)qi * T_ + kj] != 0);
      float s = valid ? sS[i * 65 + j] : NEGV;
      sS[i * 65 + j] = s;
      mx = fmaxf(mx, s);
    }
    float sum = 0.f;
    for (int j = 0; j < 64; ++j) {
      float s = sS[i * 65 + j];
      float e = (s > 0.5f * NEGV) ? __expf(s - mx) : 0.f;
      sS[i * 65 + j] = e;
      sum += e;
    }
    float inv = (sum > 0.f) ? (1.f / sum) : 0.f;
    for (int j = 0; j < 64; ++j) sP[i * 72 + j] = f2bf(sS[i * 65 + j] * inv);
  }
  __syncthreads();

  // O = P V ; P fragment shared by both tj tiles
  v8f oc[2];
  oc[0] = (v8f){0.f,0.f,0.f,0.f,0.f,0.f,0.f,0.f};
  oc[1] = (v8f){0.f,0.f,0.f,0.f,0.f,0.f,0.f,0.f};
#pragma unroll
  for (int kst = 0; kst < 2; ++kst) {
    v16bf pa  = frag_a_ld(sP  + (ti * 16) * 72 + kst * 32, 72, lane);
    v16bf vb0 = frag_b_ld(sVT + (tj0 * 16) * 72 + kst * 32, 72, lane);
    v16bf vb1 = frag_b_ld(sVT + ((tj0 + 1) * 16) * 72 + kst * 32, 72, lane);
    oc[0] = wmma_bf16(pa, vb0, oc[0]);
    oc[1] = wmma_bf16(pa, vb1, oc[1]);
  }
#pragma unroll
  for (int u = 0; u < 2; ++u)
#pragma unroll
    for (int v = 0; v < 8; ++v) {
      int row = ti * 16 + half * 8 + v;
      xout[base + (size_t)row * F_ + (tj0 + u) * 16 + cl] = f2bf(oc[u][v]);
    }
}

// ---------------- launcher ----------------
extern "C" void kernel_launch(void* const* d_in, const int* in_sizes, int n_in,
                              void* d_out, int out_size, void* d_ws, size_t ws_size,
                              hipStream_t stream) {
  (void)in_sizes; (void)n_in; (void)out_size; (void)ws_size;
  const float* query = (const float*)d_in[0];
  const float* key_  = (const float*)d_in[1];
  const float* value = (const float*)d_in[2];
  const unsigned char* mask = (const unsigned char*)d_in[3];
  const float* Wq = (const float*)d_in[4];
  const float* bq = (const float*)d_in[5];
  const float* Wk = (const float*)d_in[6];
  const float* bk = (const float*)d_in[7];
  const float* Wv = (const float*)d_in[8];
  const float* bv = (const float*)d_in[9];
  const float* Wo = (const float*)d_in[10];
  const float* bo = (const float*)d_in[11];

  char* ws = (char*)d_ws;
  const size_t WSZ = (size_t)262144;            // 512*512 elements per weight
  __bf16* wbf = (__bf16*)ws;                    // 4 packed weights: 2 MiB
  size_t off = 4 * WSZ * sizeof(__bf16);
  const size_t ACT = (size_t)B_ * TP * F_;      // padded activation element count
  __bf16* qbf = (__bf16*)(ws + off); off += ACT * sizeof(__bf16);
  __bf16* kbf = (__bf16*)(ws + off); off += ACT * sizeof(__bf16);
  __bf16* vbf = (__bf16*)(ws + off); off += ACT * sizeof(__bf16);
  __bf16* x1  = (__bf16*)(ws + off); off += ACT * sizeof(__bf16);
  __bf16* x2  = (__bf16*)(ws + off); off += ACT * sizeof(__bf16);

  // 0) weights -> bf16 fragment-packed; zero pass-2 gap rows of x2
  pack_weights<<<dim3((4 * WSZ) / 256), dim3(256), 0, stream>>>(Wq, Wk, Wv, Wo, wbf);
  zero_x2_head<<<dim3(256), dim3(256), 0, stream>>>(x2);

  // 1) projections: q0/k0/v0 (bf16, padded rows zeroed); A read once per row
  gemm_wmma<0><<<dim3(504), dim3(256), 0, stream>>>(query, nullptr, wbf + 0 * WSZ, bq, (void*)qbf);
  gemm_wmma<0><<<dim3(504), dim3(256), 0, stream>>>(key_,  nullptr, wbf + 1 * WSZ, bk, (void*)kbf);
  gemm_wmma<0><<<dim3(504), dim3(256), 0, stream>>>(value, nullptr, wbf + 2 * WSZ, bv, (void*)vbf);

  // 2) block attention: pass 0 -> x1, pass 1 -> x2 (no RMW)
  attn_block<<<dim3(NB1, H_, B_), dim3(256), 0, stream>>>(qbf, kbf, vbf, mask, x1, 0);
  attn_block<<<dim3(NB2, H_, B_), dim3(256), 0, stream>>>(qbf, kbf, vbf, mask, x2, 1);

  // 3) output projection: out = (x1 + x2)[:, :T] @ Wo + bo  (fp32 out)
  gemm_wmma<1><<<dim3(500), dim3(256), 0, stream>>>(x1, x2, wbf + 3 * WSZ, bo, d_out);
}
